// NonLocalTemporalPooling_87840671138505
// MI455X (gfx1250) — matrix operand order repaired and checked by reading, not compiled, not run on hardware
//
#include <hip/hip_runtime.h>

// ---------------------------------------------------------------------------
// NonLocalTemporalPooling — CDNA5 (gfx1250) WMMA implementation
//   BS=4, C=768, NC=384, T=8, H=W=32 -> pooled 16x16, heads=12, hd=32
//   All GEMMs + attention matmuls via v_wmma_f32_16x16x32_f16 (f16 in, f32 acc)
// ---------------------------------------------------------------------------

#define BS      4
#define CDIM    768
#define NCDIM   384
#define TDIM    8
#define NQ      1024      // 32*32 queries
#define NK      2048      // 8*16*16 keys
#define NHEADS  12
#define HD      32        // head dim == WMMA K. nice.

typedef __attribute__((ext_vector_type(16))) _Float16 v16h;
typedef __attribute__((ext_vector_type(8)))  float    v8f;

union V16H { v16h v; uint4 q[2]; _Float16 h[16]; };
union H8   { uint4 q; _Float16 h[8]; };

static __device__ inline v8f wmma16(v16h a, v16h b, v8f c) {
  // D = A(16x32 f16) * B(32x16 f16) + C(16x16 f32)
  return __builtin_amdgcn_wmma_f32_16x16x32_f16(false, a, false, b, (short)0, c,
                                                false, false);
}

// A operand (16x32, row-major source, ld = row stride in halves):
// lane<16: M=lane, K = {0..7, 16..23}; lane>=16: M=lane-16, K = {8..15, 24..31}
// p points at &mat[row][k0]; kb = (lane>=16) ? 8 : 0
static __device__ inline v16h ld_a(const _Float16* p, int kb) {
  V16H u;
  u.q[0] = *(const uint4*)(p + kb);
  u.q[1] = *(const uint4*)(p + kb + 16);
  return u.v;
}
// B operand (32x16): lane<16: N=lane, K = {0..7, 16..23}; lane>=16: N=lane-16, K = {8..15, 24..31}
// p points at row (= this lane's N column source) at k0; kb = (lane>=16) ? 8 : 0
static __device__ inline v16h ld_b(const _Float16* p, int kb) {
  V16H u;
  u.q[0] = *(const uint4*)(p + kb);
  u.q[1] = *(const uint4*)(p + kb + 16);
  return u.v;
}

static __device__ inline float redmax16(float v) {
  v = fmaxf(v, __shfl_xor(v, 1, 32));
  v = fmaxf(v, __shfl_xor(v, 2, 32));
  v = fmaxf(v, __shfl_xor(v, 4, 32));
  v = fmaxf(v, __shfl_xor(v, 8, 32));
  return v;
}
static __device__ inline float redsum16(float v) {
  v += __shfl_xor(v, 1, 32);
  v += __shfl_xor(v, 2, 32);
  v += __shfl_xor(v, 4, 32);
  v += __shfl_xor(v, 8, 32);
  return v;
}

// --------------------------- prep kernels ----------------------------------

__global__ void cvt_f2h(const float* __restrict__ src, _Float16* __restrict__ dst,
                        int n) {
  int i = blockIdx.x * blockDim.x + threadIdx.x;
  if (i < n) dst[i] = (_Float16)src[i];
}

// 2x2 spatial max-pool of x (b,c,t,32,32) -> xp[b][n][c] (c-contiguous), n=t*256+hy*16+wx
__global__ void pool_x(const float* __restrict__ x, _Float16* __restrict__ xp) {
  int i = blockIdx.x * blockDim.x + threadIdx.x;          // (b*C+c)*2048 + n
  if (i >= BS * CDIM * NK) return;
  int n  = i & (NK - 1);
  int bc = i >> 11;
  int c  = bc % CDIM;
  int b  = bc / CDIM;
  int t = n >> 8, hy = (n >> 4) & 15, wx = n & 15;
  const float* px = x + ((long)bc * TDIM + t) * 1024 + hy * 64 + wx * 2;
  float m = fmaxf(fmaxf(px[0], px[1]), fmaxf(px[32], px[33]));
  xp[((long)b * NK + n) * CDIM + c] = (_Float16)m;
}

// last = x[:, :, 7]  -> lastp[b][n][c]
__global__ void last_x(const float* __restrict__ x, _Float16* __restrict__ lp) {
  int i = blockIdx.x * blockDim.x + threadIdx.x;          // (b*C+c)*1024 + n
  if (i >= BS * CDIM * NQ) return;
  int n  = i & (NQ - 1);
  int bc = i >> 10;
  int c  = bc % CDIM;
  int b  = bc / CDIM;
  lp[((long)b * NQ + n) * CDIM + c] = (_Float16)x[((long)bc * TDIM + 7) * 1024 + n];
}

// --------------------------- K/V projection --------------------------------
// K[b][h][key][d] = wk @ xp + bk ;  Vt[b][h][d][key] = wv @ xp + bv
__global__ __launch_bounds__(128) void proj_kv(
    const _Float16* __restrict__ wkh, const _Float16* __restrict__ wvh,
    const _Float16* __restrict__ xp,
    const float* __restrict__ bk, const float* __restrict__ bv,
    _Float16* __restrict__ kh, _Float16* __restrict__ vth) {
  int lane = threadIdx.x & 31, wid = threadIdx.x >> 5;
  int ln = lane & 15, hi = lane >> 4;
  int ntile = (blockIdx.x * 4 + wid) * 16;
  int otile = blockIdx.y * 16;
  int b = blockIdx.z;
  int kbA = hi ? 8 : 0, kbB = hi ? 8 : 0;

  const _Float16* aK = wkh + (otile + ln) * CDIM;
  const _Float16* aV = wvh + (otile + ln) * CDIM;
  const _Float16* bp = xp + ((long)b * NK + ntile + ln) * CDIM;

  v8f ak = {0.f,0.f,0.f,0.f,0.f,0.f,0.f,0.f};
  v8f av = ak;
  for (int cc = 0; cc < CDIM; cc += 32) {
    v16h B = ld_b(bp + cc, kbB);
    ak = wmma16(ld_a(aK + cc, kbA), B, ak);
    av = wmma16(ld_a(aV + cc, kbA), B, av);
  }

  int n = ntile + ln;
  int head = otile >> 5;
  int dbase = (otile & 31) + 8 * hi;
  H8 ko;
#pragma unroll
  for (int r = 0; r < 8; ++r) {
    int o = otile + 8 * hi + r;
    ko.h[r] = (_Float16)(ak[r] + bk[o]);
    vth[(((long)b * NHEADS + head) * HD + dbase + r) * NK + n] =
        (_Float16)(av[r] + bv[o]);
  }
  *(uint4*)(kh + (((long)b * NHEADS + head) * NK + n) * HD + dbase) = ko.q;
}

// --------------------------- Q projection ----------------------------------
// q[b][h][n][d] = (wq @ last + bq) * hd^-0.5
__global__ __launch_bounds__(128) void proj_q(
    const _Float16* __restrict__ wqh, const _Float16* __restrict__ lp,
    const float* __restrict__ bq, _Float16* __restrict__ qh) {
  int lane = threadIdx.x & 31, wid = threadIdx.x >> 5;
  int ln = lane & 15, hi = lane >> 4;
  int ntile = (blockIdx.x * 4 + wid) * 16;
  int otile = blockIdx.y * 16;
  int b = blockIdx.z;
  int kbA = hi ? 8 : 0, kbB = hi ? 8 : 0;
  const float scale = 0.17677669529663687f;  // 32^-0.5

  const _Float16* aQ = wqh + (otile + ln) * CDIM;
  const _Float16* bp = lp + ((long)b * NQ + ntile + ln) * CDIM;

  v8f acc = {0.f,0.f,0.f,0.f,0.f,0.f,0.f,0.f};
  for (int cc = 0; cc < CDIM; cc += 32)
    acc = wmma16(ld_a(aQ + cc, kbA), ld_b(bp + cc, kbB), acc);

  int n = ntile + ln;
  int head = otile >> 5;
  int dbase = (otile & 31) + 8 * hi;
  H8 qo;
#pragma unroll
  for (int r = 0; r < 8; ++r) {
    int o = otile + 8 * hi + r;
    qo.h[r] = (_Float16)((acc[r] + bq[o]) * scale);
  }
  *(uint4*)(qh + (((long)b * NHEADS + head) * NQ + n) * HD + dbase) = qo.q;
}

// --------------------------- flash attention -------------------------------
// per wave: 16 query rows, online softmax over 64 key-blocks of 32
__global__ __launch_bounds__(128) void attn_kernel(
    const _Float16* __restrict__ qh, const _Float16* __restrict__ kh,
    const _Float16* __restrict__ vth, _Float16* __restrict__ oat) {
  __shared__ _Float16 Plds[4][16][32];
  int lane = threadIdx.x & 31, w = threadIdx.x >> 5;
  int ln = lane & 15, hi = lane >> 4;
  int task = blockIdx.x * 4 + w;          // (b*12+h)*64 + qt
  int qt = task & 63;
  int bh = task >> 6;
  int kbA = hi ? 8 : 0, kbB = hi ? 8 : 0;

  v16h Aq = ld_a(qh + ((long)bh * NQ + qt * 16 + ln) * HD, kbA);
  const _Float16* kb = kh + (long)bh * NK * HD;
  const _Float16* vb = vth + (long)bh * HD * NK;

  v8f o0 = {0.f,0.f,0.f,0.f,0.f,0.f,0.f,0.f};
  v8f o1 = o0;
  float m[8], l[8];
#pragma unroll
  for (int r = 0; r < 8; ++r) { m[r] = -__builtin_inff(); l[r] = 0.f; }

  for (int k0 = 0; k0 < NK; k0 += 32) {
    v8f z = {0.f,0.f,0.f,0.f,0.f,0.f,0.f,0.f};
    v8f s0 = wmma16(Aq, ld_b(kb + (k0 + ln) * HD, kbB), z);
    v8f s1 = wmma16(Aq, ld_b(kb + (k0 + 16 + ln) * HD, kbB), z);

#pragma unroll
    for (int r = 0; r < 8; ++r) {
      float mx = redmax16(fmaxf(s0[r], s1[r]));
      float mn = fmaxf(m[r], mx);
      float alpha = __expf(m[r] - mn);       // exp(-inf)=0 on first block
      float p0 = __expf(s0[r] - mn);
      float p1 = __expf(s1[r] - mn);
      l[r] = l[r] * alpha + redsum16(p0 + p1);
      m[r] = mn;
      o0[r] *= alpha;
      o1[r] *= alpha;
      Plds[w][r + 8 * hi][ln]      = (_Float16)p0;   // D-layout -> row-major
      Plds[w][r + 8 * hi][16 + ln] = (_Float16)p1;
    }
    asm volatile("s_wait_dscnt 0" ::: "memory");      // same-wave LDS RAW
    V16H ap;                                          // reload in A-layout
    ap.q[0] = *(const uint4*)&Plds[w][ln][kbA];
    ap.q[1] = *(const uint4*)&Plds[w][ln][kbA + 16];

    o0 = wmma16(ap.v, ld_b(vb + (long)ln * NK + k0, kbB), o0);
    o1 = wmma16(ap.v, ld_b(vb + (long)(16 + ln) * NK + k0, kbB), o1);
  }

  int b = bh / NHEADS, h = bh % NHEADS;
#pragma unroll
  for (int r = 0; r < 8; ++r) {
    float inv = 1.0f / l[r];
    int q = qt * 16 + r + 8 * hi;
    long base = ((long)b * NQ + q) * NCDIM + h * HD;
    oat[base + ln]      = (_Float16)(o0[r] * inv);
    oat[base + 16 + ln] = (_Float16)(o1[r] * inv);
  }
}

// --------------------------- output projection -----------------------------
// out[b][co][n] = wo @ o_attn + bo + last   (fp32 out)
__global__ __launch_bounds__(128) void proj_out(
    const _Float16* __restrict__ woh, const _Float16* __restrict__ oat,
    const float* __restrict__ bo, const float* __restrict__ x,
    float* __restrict__ out) {
  int lane = threadIdx.x & 31, wid = threadIdx.x >> 5;
  int ln = lane & 15, hi = lane >> 4;
  int ntile = (blockIdx.x * 4 + wid) * 16;
  int cot = blockIdx.y * 16;
  int b = blockIdx.z;
  int kbA = hi ? 8 : 0, kbB = hi ? 8 : 0;

  const _Float16* ap = woh + (cot + ln) * NCDIM;
  const _Float16* bp = oat + ((long)b * NQ + ntile + ln) * NCDIM;

  v8f acc = {0.f,0.f,0.f,0.f,0.f,0.f,0.f,0.f};
  for (int nc = 0; nc < NCDIM; nc += 32)
    acc = wmma16(ld_a(ap + nc, kbA), ld_b(bp + nc, kbB), acc);

  int n = ntile + ln;
#pragma unroll
  for (int r = 0; r < 8; ++r) {
    int co = cot + 8 * hi + r;
    out[((long)b * CDIM + co) * NQ + n] =
        acc[r] + bo[co] + x[(((long)b * CDIM + co) * TDIM + 7) * NQ + n];
  }
}

// ----------------------------- launcher ------------------------------------

extern "C" void kernel_launch(void* const* d_in, const int* in_sizes, int n_in,
                              void* d_out, int out_size, void* d_ws, size_t ws_size,
                              hipStream_t stream) {
  const float* x  = (const float*)d_in[0];
  const float* wq = (const float*)d_in[1];
  const float* bq = (const float*)d_in[2];
  const float* wk = (const float*)d_in[3];
  const float* bk = (const float*)d_in[4];
  const float* wv = (const float*)d_in[5];
  const float* bv = (const float*)d_in[6];
  const float* wo = (const float*)d_in[7];
  const float* bo = (const float*)d_in[8];
  float* out = (float*)d_out;

  char* ws = (char*)d_ws;
  size_t off = 0;
  auto alloc = [&](size_t bytes) -> void* {
    void* p = ws + off;
    off += (bytes + 255) & ~(size_t)255;
    return p;
  };
  const int WSZ = NCDIM * CDIM;                 // 294912 (wq/wk/wv/wo all same count)
  _Float16* wq_h = (_Float16*)alloc((size_t)WSZ * 2);
  _Float16* wk_h = (_Float16*)alloc((size_t)WSZ * 2);
  _Float16* wv_h = (_Float16*)alloc((size_t)WSZ * 2);
  _Float16* wo_h = (_Float16*)alloc((size_t)WSZ * 2);
  _Float16* xp_h = (_Float16*)alloc((size_t)BS * NK * CDIM * 2);
  _Float16* lp_h = (_Float16*)alloc((size_t)BS * NQ * CDIM * 2);
  _Float16* q_h  = (_Float16*)alloc((size_t)BS * NHEADS * NQ * HD * 2);
  _Float16* k_h  = (_Float16*)alloc((size_t)BS * NHEADS * NK * HD * 2);
  _Float16* vt_h = (_Float16*)alloc((size_t)BS * NHEADS * HD * NK * 2);
  _Float16* oa_h = (_Float16*)alloc((size_t)BS * NQ * NCDIM * 2);
  (void)ws_size; (void)n_in; (void)in_sizes; (void)out_size;

  // weights -> f16
  int wb = (WSZ + 255) / 256;
  cvt_f2h<<<wb, 256, 0, stream>>>(wq, wq_h, WSZ);
  cvt_f2h<<<wb, 256, 0, stream>>>(wk, wk_h, WSZ);
  cvt_f2h<<<wb, 256, 0, stream>>>(wv, wv_h, WSZ);
  cvt_f2h<<<wb, 256, 0, stream>>>(wo, wo_h, WSZ);

  // pool + last extraction (transposed, f16)
  pool_x<<<(BS * CDIM * NK + 255) / 256, 256, 0, stream>>>(x, xp_h);
  last_x<<<(BS * CDIM * NQ + 255) / 256, 256, 0, stream>>>(x, lp_h);

  // projections
  proj_kv<<<dim3(NK / 64, NCDIM / 16, BS), 128, 0, stream>>>(
      wk_h, wv_h, xp_h, bk, bv, k_h, vt_h);
  proj_q<<<dim3(NQ / 64, NCDIM / 16, BS), 128, 0, stream>>>(wq_h, lp_h, bq, q_h);

  // attention: 4 waves/block, one 16-query tile per wave
  attn_kernel<<<BS * NHEADS * (NQ / 16) / 4, 128, 0, stream>>>(q_h, k_h, vt_h, oa_h);

  // output projection + bias + residual
  proj_out<<<dim3(NQ / 64, CDIM / 16, BS), 128, 0, stream>>>(wo_h, oa_h, bo, x, out);
}
